// ByteEncoder_57724360458272
// MI455X (gfx1250) — matrix operand order repaired and checked by reading, not compiled, not run on hardware
//
#include <hip/hip_runtime.h>
#include <hip/hip_bf16.h>
#include <math.h>

// ---------------------------------------------------------------------------
// Types for CDNA5 WMMA (wave32): A/B are 16 bf16 per lane, C/D are 8 f32.
// ---------------------------------------------------------------------------
typedef __bf16 bf16_t;
typedef bf16_t v16bf __attribute__((ext_vector_type(16)));
typedef float  v8f   __attribute__((ext_vector_type(8)));

union V16U { v16bf v; uint4 q[2]; };
union US8U { unsigned short s[8]; uint4 q; };

__device__ __forceinline__ unsigned short f32_to_bf16u(float f) {
  unsigned int u = __float_as_uint(f);
  u += 0x7FFFu + ((u >> 16) & 1u);      // round-to-nearest-even
  return (unsigned short)(u >> 16);
}

// Async global->LDS copy of 16 bytes (per active lane). gfx1250 TDM-adjacent
// path: no VGPR round trip, tracked by ASYNCcnt. LDS offset = low 32 bits of
// the generic pointer (ISA 10.2: LDS aperture maps addr[31:0] -> LDS offset).
__device__ __forceinline__ void async_copy_b128(void* lds_ptr, const void* gptr) {
  unsigned loff = (unsigned)(size_t)lds_ptr;
  asm volatile("global_load_async_to_lds_b128 %0, %1, off"
               :: "v"(loff), "v"((unsigned long long)(size_t)gptr)
               : "memory");
}
__device__ __forceinline__ void async_wait0() {
  asm volatile("s_wait_asynccnt 0x0" ::: "memory");
}

#define D_MODEL 1024
#define T_SEQ   8192
#define BT_TOT  32768   // B*T
#define N2_TOT  8192    // B*T/4

// ---------------------------------------------------------------------------
// K0: embedding gather, f32 -> bf16, h[row, d]
// ---------------------------------------------------------------------------
__global__ __launch_bounds__(256) void embed_gather(
    const int* __restrict__ x, const float* __restrict__ embed,
    unsigned short* __restrict__ h) {
  int row = blockIdx.x;
  int tok = x[row];
  const float* src = embed + (size_t)tok * D_MODEL;
  unsigned short* dst = h + (size_t)row * D_MODEL;
#pragma unroll
  for (int i = 0; i < 4; ++i) {
    int d = threadIdx.x + i * 256;
    dst[d] = f32_to_bf16u(src[d]);
  }
}

// ---------------------------------------------------------------------------
// K1: repack conv weights [CO, CI, K] f32 -> [CO][k*CI + ci] bf16
// (K==1 => plain f32->bf16 matrix convert, used for b_w / c_w too)
// ---------------------------------------------------------------------------
__global__ __launch_bounds__(256) void repack_conv_w(
    const float* __restrict__ w, unsigned short* __restrict__ out,
    int CI, int K, int n) {
  int idx = blockIdx.x * 256 + threadIdx.x;
  if (idx >= n) return;
  int cik = CI * K;
  int co = idx / cik;
  int rem = idx - co * cik;
  int ci = rem / K;
  int k  = rem - ci * K;
  out[(size_t)co * cik + k * CI + ci] = f32_to_bf16u(w[idx]);
}

__global__ __launch_bounds__(256) void pack_stem_bias(
    const float* __restrict__ b1, const float* __restrict__ b2,
    const float* __restrict__ b4, const float* __restrict__ b8,
    float* __restrict__ out) {
  int i = blockIdx.x * 256 + threadIdx.x;  // 0..1023
  float v = (i < 256) ? b1[i] : (i < 512) ? b2[i - 256]
          : (i < 768) ? b4[i - 512] : b8[i - 768];
  out[i] = v;
}

// ---------------------------------------------------------------------------
// K2: multi-scale conv stem as WMMA GEMM.
// Output tile 128 co x 64 t, 8 waves (wave: 16 co x 64 t = 4 wmma tiles).
// One 72-row halo tile [tBase-4, tBase+68) x 32 ci is staged per ci-step and
// shared by ALL taps (shift just re-indexes LDS rows). Ping-pong buffers +
// async global->LDS staging: one barrier per ci-step.
// Epilogue: +bias, exact GELU, -> bf16 h_multi[t, co].
// ---------------------------------------------------------------------------
__global__ __launch_bounds__(256) void stem_gemm(
    const unsigned short* __restrict__ h,       // [BT, 1024] bf16
    const unsigned short* __restrict__ wpack,   // packed conv weights
    const float* __restrict__ sbias,            // concat biases [1024]
    unsigned short* __restrict__ hmulti) {      // [BT, 1024] bf16
  __shared__ __align__(32) unsigned short Bt[2][72 * 32];  // 2 x 4.5KB halo tiles

  const int tid  = threadIdx.x;
  const int wave = tid >> 5;
  const int lane = tid & 31;
  const int lm = lane & 15, lh = lane >> 4;

  const int tBase  = blockIdx.x * 64;
  const int coBase = blockIdx.y * 128;
  const int conv = coBase >> 8;          // 0..3
  const int taps = 1 << conv;            // 1,2,4,8
  const int pad  = (conv == 0) ? 0 : (taps >> 1);
  const int lda  = taps * D_MODEL;
  const int convOff[4] = {0, 262144, 786432, 1835008};  // ushort offsets
  const unsigned short* wconv = wpack + convOff[conv];
  const int coInConv = coBase & 255;
  const int bStart = tBase & ~(T_SEQ - 1);   // tile never crosses batch edge

  v8f acc[4] = {};

  // stage 72t x 32ci halo tile for ci-block c into buffer `buf`
  auto stageStem = [&](int buf, int c) {
    for (int ch = tid; ch < 288; ch += 256) {   // 72 rows x 4 chunks
      int tl = ch >> 2;
      int cc = (ch & 3) * 8;
      int tg = tBase - 4 + tl;
      unsigned short* lp = &Bt[buf][tl * 32 + cc];
      if (tg >= bStart && tg < bStart + T_SEQ) {
        async_copy_b128(lp, h + (size_t)tg * D_MODEL + c + cc);
      } else {
        *(uint4*)lp = make_uint4(0u, 0u, 0u, 0u);   // zero halo (DScnt)
      }
    }
  };

  stageStem(0, 0);
  async_wait0();
  __syncthreads();

  int p = 0;
  for (int c = 0; c < D_MODEL; c += 32) {
    if (c + 32 < D_MODEL) stageStem(p ^ 1, c + 32);   // prefetch next ci-block
    for (int k = 0; k < taps; ++k) {
      const int shift = k - pad;
      V16U a;
      {
        const unsigned short* ap =
            wconv + (size_t)(coInConv + wave * 16 + lm) * lda
                  + k * D_MODEL + c + (lh ? 8 : 0);
        a.q[0] = *(const uint4*)ap;
        a.q[1] = *(const uint4*)(ap + 16);
      }
#pragma unroll
      for (int n = 0; n < 4; ++n) {
        const unsigned short* bp =
            &Bt[p][(n * 16 + lm + shift + 4) * 32 + lh * 8];
        V16U b;
        b.q[0] = *(const uint4*)bp;
        b.q[1] = *(const uint4*)(bp + 16);
        acc[n] = __builtin_amdgcn_wmma_f32_16x16x32_bf16(
            false, a.v, false, b.v, (short)0, acc[n], false, false);
      }
    }
    async_wait0();          // next-step async stages landed
    __syncthreads();        // everyone done reading Bt[p]
    p ^= 1;
  }

  // ---- epilogue: bias + exact GELU -> bf16 [t, co] ----
  const int co0 = coBase + wave * 16 + (lh ? 8 : 0);
#pragma unroll
  for (int n = 0; n < 4; ++n) {
    int t = tBase + n * 16 + lm;
    US8U o;
#pragma unroll
    for (int r = 0; r < 8; ++r) {
      float xv = acc[n][r] + sbias[co0 + r];
      float g = 0.5f * xv * (1.0f + erff(xv * 0.70710678118654752f));
      o.s[r] = f32_to_bf16u(g);
    }
    *(uint4*)(hmulti + (size_t)t * D_MODEL + co0) = o.q;
  }
}

// ---------------------------------------------------------------------------
// K3: generic WMMA GEMM  out[t, co] = sum_k A[co,k] * B[t*bRowStride + k]
// M = 1024 fixed. Ping-pong LDS + async global->LDS staging.
// Modes: 1 = +bias (down conv -> f32)
//        2 = (1-sigmoid(raw[co]))*(+bias)   (LRU values)
//        3 = +bias + residual[t,co]          (C proj)
// ---------------------------------------------------------------------------
__global__ __launch_bounds__(256) void gemm_bf16(
    const unsigned short* __restrict__ A,     // [1024, Ktot] bf16
    const unsigned short* __restrict__ Bsrc,  // col t = Bsrc + t*bRowStride
    const float* __restrict__ bias,
    const float* __restrict__ extra,          // mode2: log_lambda_raw; mode3: residual
    float* __restrict__ outF,
    int Ktot, int bRowStride, int mode) {
  __shared__ __align__(32) unsigned short Bt[2][64 * 32];

  const int tid  = threadIdx.x;
  const int wave = tid >> 5;
  const int lane = tid & 31;
  const int lm = lane & 15, lh = lane >> 4;
  const int tBase  = blockIdx.x * 64;
  const int coBase = blockIdx.y * 128;

  const int tl = tid >> 2;            // 0..63
  const int cc = (tid & 3) * 8;       // 0,8,16,24

  v8f acc[4] = {};

  auto stageB = [&](int buf, int c) {
    async_copy_b128(&Bt[buf][tl * 32 + cc],
                    Bsrc + (size_t)(tBase + tl) * bRowStride + c + cc);
  };

  stageB(0, 0);
  async_wait0();
  __syncthreads();

  int p = 0;
  for (int c = 0; c < Ktot; c += 32) {
    if (c + 32 < Ktot) stageB(p ^ 1, c + 32);
    V16U a;
    {
      const unsigned short* ap =
          A + (size_t)(coBase + wave * 16 + lm) * Ktot + c + (lh ? 8 : 0);
      a.q[0] = *(const uint4*)ap;
      a.q[1] = *(const uint4*)(ap + 16);
    }
#pragma unroll
    for (int n = 0; n < 4; ++n) {
      const unsigned short* bp = &Bt[p][(n * 16 + lm) * 32 + lh * 8];
      V16U b;
      b.q[0] = *(const uint4*)bp;
      b.q[1] = *(const uint4*)(bp + 16);
      acc[n] = __builtin_amdgcn_wmma_f32_16x16x32_bf16(
          false, a.v, false, b.v, (short)0, acc[n], false, false);
    }
    async_wait0();
    __syncthreads();
    p ^= 1;
  }

  const int co0 = coBase + wave * 16 + (lh ? 8 : 0);
#pragma unroll
  for (int n = 0; n < 4; ++n) {
    int t = tBase + n * 16 + lm;
    float r[8];
#pragma unroll
    for (int i = 0; i < 8; ++i) r[i] = acc[n][i] + bias[co0 + i];
    if (mode == 2) {
#pragma unroll
      for (int i = 0; i < 8; ++i) {
        float lam = 1.0f / (1.0f + expf(-extra[co0 + i]));
        r[i] *= (1.0f - lam);
      }
    } else if (mode == 3) {
#pragma unroll
      for (int i = 0; i < 8; ++i) r[i] += extra[(size_t)t * D_MODEL + co0 + i];
    }
    float* op = outF + (size_t)t * D_MODEL + co0;
    *(float4*)(op)     = make_float4(r[0], r[1], r[2], r[3]);
    *(float4*)(op + 4) = make_float4(r[4], r[5], r[6], r[7]);
  }
}

// ---------------------------------------------------------------------------
// K4: LayerNorm over last dim (1024). Optionally emit bf16 copy for GEMM B.
// ---------------------------------------------------------------------------
__global__ __launch_bounds__(256) void layernorm_k(
    const float* __restrict__ in, const float* __restrict__ g,
    const float* __restrict__ bta, float* __restrict__ outF,
    unsigned short* __restrict__ outBF) {
  __shared__ float sh[256];
  const int row = blockIdx.x;
  const float* src = in + (size_t)row * D_MODEL;
  float vals[4], s = 0.f, sq = 0.f;
#pragma unroll
  for (int i = 0; i < 4; ++i) {
    float v = src[threadIdx.x + i * 256];
    vals[i] = v; s += v; sq += v * v;
  }
  sh[threadIdx.x] = s; __syncthreads();
  for (int o = 128; o > 0; o >>= 1) {
    if (threadIdx.x < o) sh[threadIdx.x] += sh[threadIdx.x + o];
    __syncthreads();
  }
  float mean = sh[0] * (1.0f / D_MODEL);
  __syncthreads();
  sh[threadIdx.x] = sq; __syncthreads();
  for (int o = 128; o > 0; o >>= 1) {
    if (threadIdx.x < o) sh[threadIdx.x] += sh[threadIdx.x + o];
    __syncthreads();
  }
  float var = sh[0] * (1.0f / D_MODEL) - mean * mean;
  float rstd = rsqrtf(var + 1e-5f);
#pragma unroll
  for (int i = 0; i < 4; ++i) {
    int d = threadIdx.x + i * 256;
    float y = (vals[i] - mean) * rstd * g[d] + bta[d];
    outF[(size_t)row * D_MODEL + d] = y;
    if (outBF) outBF[(size_t)row * D_MODEL + d] = f32_to_bf16u(y);
  }
}

// ---------------------------------------------------------------------------
// K5: per-channel linear recurrence h[t] = lam*h[t-1] + values[t].
// values already scaled by (1-lam). Loads prefetched 8-deep; only the FMA
// chain is serial. grid = (B=4, D/256=4).
// ---------------------------------------------------------------------------
__global__ __launch_bounds__(256) void lru_scan(
    const float* __restrict__ values, const float* __restrict__ raw,
    unsigned short* __restrict__ hallbf) {
  const int b = blockIdx.x;
  const int d = blockIdx.y * 256 + threadIdx.x;
  const float lam = 1.0f / (1.0f + expf(-raw[d]));
  const float* v = values + (size_t)b * 2048 * D_MODEL + d;
  unsigned short* o = hallbf + (size_t)b * 2048 * D_MODEL + d;
  float hstate = 0.0f;
  for (int t = 0; t < 2048; t += 8) {
    float x[8];
#pragma unroll
    for (int i = 0; i < 8; ++i) x[i] = v[(size_t)(t + i) * D_MODEL];
#pragma unroll
    for (int i = 0; i < 8; ++i) {
      hstate = fmaf(lam, hstate, x[i]);
      o[(size_t)(t + i) * D_MODEL] = f32_to_bf16u(hstate);
    }
  }
}

// ---------------------------------------------------------------------------
// Launcher. Workspace layout (bytes), peak ~239 MB with aliasing:
//   [0,64M)      h bf16            -> later aliased by values(f32,33.5M)+hall(bf16,16.8M)
//   [64M,128M)   h_multi bf16      -> later aliased by tmp f32 (33.5M)
//   [128M,148M)  packed weights + stem bias
//   [~148M..)    h_down f32, h_s f32, h_s bf16
// ---------------------------------------------------------------------------
extern "C" void kernel_launch(void* const* d_in, const int* in_sizes, int n_in,
                              void* d_out, int out_size, void* d_ws, size_t ws_size,
                              hipStream_t stream) {
  (void)in_sizes; (void)n_in; (void)out_size; (void)ws_size;
  const int*   x    = (const int*)  d_in[0];
  const float* emb  = (const float*)d_in[1];
  const float* c1w  = (const float*)d_in[2];
  const float* c1b  = (const float*)d_in[3];
  const float* c2w  = (const float*)d_in[4];
  const float* c2b  = (const float*)d_in[5];
  const float* c4w  = (const float*)d_in[6];
  const float* c4b  = (const float*)d_in[7];
  const float* c8w  = (const float*)d_in[8];
  const float* c8b  = (const float*)d_in[9];
  const float* dww  = (const float*)d_in[10];
  const float* dwb  = (const float*)d_in[11];
  const float* slw  = (const float*)d_in[12];
  const float* slb  = (const float*)d_in[13];
  const float* raw  = (const float*)d_in[14];
  const float* bw   = (const float*)d_in[15];
  const float* bb   = (const float*)d_in[16];
  const float* cw   = (const float*)d_in[17];
  const float* cb   = (const float*)d_in[18];
  const float* llw  = (const float*)d_in[19];
  const float* llb  = (const float*)d_in[20];

  char* ws = (char*)d_ws;
  unsigned short* h      = (unsigned short*)(ws + 0);
  unsigned short* hmulti = (unsigned short*)(ws + 67108864);
  unsigned short* wpack  = (unsigned short*)(ws + 134217728);
  float*          sbias  = (float*)(ws + 154664960);
  float*          hdown  = (float*)(ws + 155189248);
  float*          hs     = (float*)(ws + 188743680);
  unsigned short* hsbf   = (unsigned short*)(ws + 222298112);
  float*          values = (float*)(ws + 0);                    // alias h
  unsigned short* hallbf = (unsigned short*)(ws + 33554432);    // alias h tail
  float*          tmp    = (float*)(ws + 67108864);             // alias hmulti

  // packed weight offsets in ushort elements
  unsigned short* w1 = wpack + 0;
  unsigned short* w2 = wpack + 262144;
  unsigned short* w4 = wpack + 786432;
  unsigned short* w8 = wpack + 1835008;
  unsigned short* wd = wpack + 3932160;
  unsigned short* wbp = wpack + 8126464;
  unsigned short* wcp = wpack + 9175040;

  // K0: embed gather -> bf16
  embed_gather<<<BT_TOT, 256, 0, stream>>>(x, emb, h);

  // K1: weight repacks (f32 -> bf16, tap-major K layout)
  auto blocks = [](int n) { return (n + 255) / 256; };
  repack_conv_w<<<blocks(256 * 1024 * 1), 256, 0, stream>>>(c1w, w1, 1024, 1, 256 * 1024 * 1);
  repack_conv_w<<<blocks(256 * 1024 * 2), 256, 0, stream>>>(c2w, w2, 1024, 2, 256 * 1024 * 2);
  repack_conv_w<<<blocks(256 * 1024 * 4), 256, 0, stream>>>(c4w, w4, 1024, 4, 256 * 1024 * 4);
  repack_conv_w<<<blocks(256 * 1024 * 8), 256, 0, stream>>>(c8w, w8, 1024, 8, 256 * 1024 * 8);
  repack_conv_w<<<blocks(1024 * 1024 * 4), 256, 0, stream>>>(dww, wd, 1024, 4, 1024 * 1024 * 4);
  repack_conv_w<<<blocks(1024 * 1024), 256, 0, stream>>>(bw, wbp, 1024, 1, 1024 * 1024);
  repack_conv_w<<<blocks(1024 * 1024), 256, 0, stream>>>(cw, wcp, 1024, 1, 1024 * 1024);
  pack_stem_bias<<<4, 256, 0, stream>>>(c1b, c2b, c4b, c8b, sbias);

  // K2: multi-scale conv stem (WMMA) -> gelu -> h_multi bf16 [BT, 1024]
  stem_gemm<<<dim3(BT_TOT / 64, 8), 256, 0, stream>>>(h, wpack, sbias, hmulti);

  // K3a: down conv as K=4096 GEMM -> h_down f32 [N2, 1024]
  gemm_bf16<<<dim3(N2_TOT / 64, 8), 256, 0, stream>>>(
      wd, hmulti, dwb, nullptr, hdown, 4096, 4096, 1);

  // K4a: stem LayerNorm -> h_s (f32 + bf16)
  layernorm_k<<<N2_TOT, 256, 0, stream>>>(hdown, slw, slb, hs, hsbf);

  // K3b: B projection, fused (1-lam) scaling -> values f32
  gemm_bf16<<<dim3(N2_TOT / 64, 8), 256, 0, stream>>>(
      wbp, hsbf, bb, raw, values, 1024, 1024, 2);

  // K5: linear recurrence -> h_all bf16
  lru_scan<<<dim3(4, 4), 256, 0, stream>>>(values, raw, hallbf);

  // K3c: C projection, fused residual (+h_s) -> tmp f32
  gemm_bf16<<<dim3(N2_TOT / 64, 8), 256, 0, stream>>>(
      wcp, hallbf, cb, hs, tmp, 1024, 1024, 3);

  // K4b: final LayerNorm -> d_out f32 [B, T/4, D]
  layernorm_k<<<N2_TOT, 256, 0, stream>>>(tmp, llw, llb, (float*)d_out, nullptr);
}
